// Mamba_40046275068536
// MI455X (gfx1250) — matrix operand order repaired and checked by reading, not compiled, not run on hardware
//
#include <hip/hip_runtime.h>
#include <hip/hip_bf16.h>
#include <math.h>
#include <stdint.h>

// ---- problem constants (from reference) ----
#define D_MODEL  1024
#define D_STATE  16
#define D_CONV   4
#define D_INNER  2048
#define DT_RANK  64
#define BATCH    2
#define SEQLEN   2048
#define NTOK     (BATCH * SEQLEN)          // 4096 tokens
#define XPROJ_N  (DT_RANK + 2 * D_STATE)   // 96

typedef __attribute__((ext_vector_type(16))) __bf16 v16bf;
typedef __attribute__((ext_vector_type(8)))  __bf16 v8bf;
typedef __attribute__((ext_vector_type(4)))  __bf16 v4bf;
typedef __attribute__((ext_vector_type(8)))  float  v8f;
typedef __attribute__((ext_vector_type(4)))  float  v4f;
typedef __attribute__((ext_vector_type(4)))  unsigned int v4u;
typedef __attribute__((ext_vector_type(8)))  int    v8i;
typedef __attribute__((ext_vector_type(4)))  int    v4i;

__device__ __forceinline__ float silu_f(float x)     { return x / (1.0f + __expf(-x)); }
__device__ __forceinline__ float softplus_f(float x) { return (x > 20.0f) ? x : log1pf(__expf(x)); }

// -----------------------------------------------------------------------------
// TDM: DMA a 2D bf16 tile (tile_k x tile_rows, row stride = stride_elems) from
// global memory into LDS, with LDS padding of 4 dwords after every 16 dwords
// (row of 32 bf16 -> effective LDS pitch 40 bf16 = 20 banks).
// D# built per CDNA5 ISA 8.3/8.4 (group0: count/lds/global/type, group1:
// data_size, pad, dims, strides). Tiles are in-bounds by construction, so
// tensor dims == tile dims. Groups 2/3 + trailing v8i are zero (2D, no gather).
// -----------------------------------------------------------------------------
__device__ __forceinline__ void tdm_load_tile_2d(const __bf16* gsrc, void* lds_dst,
                                                 unsigned tile_k, unsigned tile_rows,
                                                 unsigned stride_elems)
{
    const unsigned long long ga = (unsigned long long)(uintptr_t)gsrc;
    const unsigned lds_off = (unsigned)(uintptr_t)lds_dst;  // LDS aperture: addr[31:0]

    v4u g0;
    g0[0] = 1u;                                              // count=1, is_restore=0
    g0[1] = lds_off;                                         // lds_addr
    g0[2] = (unsigned)(ga & 0xFFFFFFFFull);                  // global_addr[31:0]
    g0[3] = (unsigned)((ga >> 32) & 0x01FFFFFFull)           // global_addr[56:32]
          | (2u << 30);                                      // type=2 ("image")

    v8i g1;
    g1[0] = (int)((1u << 16)                                 // data_size=1 -> 2 bytes
          | (1u << 20)                                       // pad_enable
          | (3u << 22)                                       // pad_interval: 16 dwords
          | (3u << 25));                                     // pad_amount: 4 dwords
    g1[1] = (int)((tile_k & 0xFFFFu) << 16);                 // tensor_dim0[15:0] (abar=0)
    g1[2] = (int)(((tile_k >> 16) & 0xFFFFu)                 // tensor_dim0[31:16]
          | ((tile_rows & 0xFFFFu) << 16));                  // tensor_dim1[15:0]
    g1[3] = (int)(((tile_rows >> 16) & 0xFFFFu)              // tensor_dim1[31:16]
          | ((tile_k & 0xFFFFu) << 16));                     // tile_dim0
    g1[4] = (int)(tile_rows & 0xFFFFu);                      // tile_dim1 (tile_dim2=0 -> 2D)
    g1[5] = (int)stride_elems;                               // tensor_dim0_stride[31:0]
    g1[6] = 0;                                               // stride hi / dim1_stride lo
    g1[7] = 0;

    const v4i gz4 = {0, 0, 0, 0};                            // groups 2/3 unused (2D)
    const v8i gz8 = {0, 0, 0, 0, 0, 0, 0, 0};                // trailing group (unused)
    __builtin_amdgcn_tensor_load_to_lds(g0, g1, gz4, gz4, gz8, 0);
}

// -----------------------------------------------------------------------------
// f32 -> bf16 conversion pass (4 elements / thread, n always a multiple of 4)
// -----------------------------------------------------------------------------
__global__ void cvt_f32_bf16(const float* __restrict__ in, __bf16* __restrict__ out, int n)
{
    const int i = (blockIdx.x * blockDim.x + threadIdx.x) * 4;
    if (i >= n) return;
    const v4f v = *(const v4f*)(in + i);
    v4bf o;
    o[0] = (__bf16)v[0]; o[1] = (__bf16)v[1]; o[2] = (__bf16)v[2]; o[3] = (__bf16)v[3];
    *(v4bf*)(out + i) = o;
}

// -----------------------------------------------------------------------------
// GEMM: C[M,N] = A[M,K] * W[N,K]^T, bf16 operands, f32 accumulate via
// v_wmma_f32_16x16x32_bf16. Block = 128 threads (4 waves): 64(M) x NSUB*16(N).
// Both A and B tiles are DMA'd into double-buffered LDS by the TDM (wave 0
// issues tensor_load_to_lds; s_wait_tensorcnt + barrier publishes the tile,
// DMA of step i+1 overlaps WMMA of step i).
// EPI: 0 = plain store, 1 = softplus(x + bias[col]).
// -----------------------------------------------------------------------------
template <int NSUB, int EPI>
__global__ void gemm_bf16_wmma(const __bf16* __restrict__ A,
                               const __bf16* __restrict__ W,
                               const float*  __restrict__ bias,
                               float* __restrict__ C,
                               __bf16* __restrict__ Cbf,
                               int K, int lda, int ldw, int ldc, int n_base)
{
    // rows of 32 bf16 + 8 bf16 TDM pad -> pitch 40 bf16 (20 banks)
    __shared__ alignas(16) __bf16 As[2][64][40];
    __shared__ alignas(16) __bf16 Bs[2][NSUB * 16][40];

    const int lane     = threadIdx.x & 31;
    const int wave     = threadIdx.x >> 5;           // 0..3
    const int m_blk    = blockIdx.y * 64;
    const int n_origin = n_base + blockIdx.x * (NSUB * 16);
    const int mrow     = lane & 15;
    const int khalf    = lane >> 4;

    v8f acc[NSUB];
    #pragma unroll
    for (int j = 0; j < NSUB; ++j)
        #pragma unroll
        for (int r = 0; r < 8; ++r)
            acc[j][r] = 0.0f;

    const int nk = K / 32;
    if (wave == 0) {
        tdm_load_tile_2d(A + (size_t)m_blk * lda,    &As[0][0][0], 32, 64,        lda);
        tdm_load_tile_2d(W + (size_t)n_origin * ldw, &Bs[0][0][0], 32, NSUB * 16, ldw);
    }

    for (int it = 0; it < nk; ++it) {
        const int buf = it & 1;
        if (wave == 0)
            __builtin_amdgcn_s_wait_tensorcnt(0);    // tile `it` has landed
        __syncthreads();                             // publish tile; fences buffer reuse
        if (wave == 0 && (it + 1) < nk) {
            const int k0n = (it + 1) * 32;
            tdm_load_tile_2d(A + (size_t)m_blk * lda + k0n,    &As[buf ^ 1][0][0], 32, 64,        lda);
            tdm_load_tile_2d(W + (size_t)n_origin * ldw + k0n, &Bs[buf ^ 1][0][0], 32, NSUB * 16, ldw);
        }

        // ---- A fragment: lane (khalf*16+m): K = khalf*8+[0..7] and 16+khalf*8+[0..7]
        const __bf16* arow = &As[buf][wave * 16 + mrow][khalf * 8];
        const v8bf alo = *(const v8bf*)(arow);
        const v8bf ahi = *(const v8bf*)(arow + 16);
        const v16bf afrag = __builtin_shufflevector(alo, ahi,
            0, 1, 2, 3, 4, 5, 6, 7, 8, 9, 10, 11, 12, 13, 14, 15);

        // ---- NSUB wmma with B fragments: n = lane&15, K = khalf*8+[0..7] and 16+khalf*8+[0..7]
        #pragma unroll
        for (int j = 0; j < NSUB; ++j) {
            const v8bf blo = *(const v8bf*)&Bs[buf][j * 16 + mrow][khalf * 8];
            const v8bf bhi = *(const v8bf*)&Bs[buf][j * 16 + mrow][khalf * 8 + 16];
            const v16bf bfrag = __builtin_shufflevector(blo, bhi,
                0, 1, 2, 3, 4, 5, 6, 7, 8, 9, 10, 11, 12, 13, 14, 15);
            acc[j] = __builtin_amdgcn_wmma_f32_16x16x32_bf16(
                false, afrag, false, bfrag, (short)0, acc[j], false, false);
        }
    }

    // ---- store: acc element r -> row m_blk + wave*16 + r + 8*khalf, col n_origin + j*16 + mrow
    #pragma unroll
    for (int j = 0; j < NSUB; ++j) {
        const int col = n_origin + j * 16 + mrow;
        #pragma unroll
        for (int r = 0; r < 8; ++r) {
            const int row = m_blk + wave * 16 + r + 8 * khalf;
            float v = acc[j][r];
            if (EPI == 1) v = softplus_f(v + bias[col]);
            C[(size_t)row * ldc + col] = v;
            if (Cbf) Cbf[(size_t)row * ldc + col] = (__bf16)v;
        }
    }
}

// -----------------------------------------------------------------------------
// Causal depthwise conv (width 4) + bias + SiLU; writes f32 + bf16 mirror.
// -----------------------------------------------------------------------------
__global__ void conv_silu_kernel(const float* __restrict__ xz,
                                 const float* __restrict__ conv_w,
                                 const float* __restrict__ conv_b,
                                 float* __restrict__ xc,
                                 __bf16* __restrict__ xc_bf)
{
    const int idx = blockIdx.x * blockDim.x + threadIdx.x;
    if (idx >= NTOK * D_INNER) return;
    const int d   = idx % D_INNER;
    const int tok = idx / D_INNER;
    const int b   = tok / SEQLEN;
    const int l   = tok % SEQLEN;

    float acc = conv_b[d];
    #pragma unroll
    for (int w = 0; w < D_CONV; ++w) {
        const int ll = l - (D_CONV - 1) + w;
        if (ll >= 0)
            acc += conv_w[d * D_CONV + w] *
                   xz[((size_t)(b * SEQLEN + ll)) * (2 * D_INNER) + d];
    }
    const float v = silu_f(acc);
    xc[(size_t)tok * D_INNER + d]    = v;
    xc_bf[(size_t)tok * D_INNER + d] = (__bf16)v;
}

// -----------------------------------------------------------------------------
// Selective scan: 16 lanes per (b,d) channel, one lane per state index n.
// -----------------------------------------------------------------------------
__global__ void scan_kernel(const float* __restrict__ xc,     // u     (tok, D_INNER)
                            const float* __restrict__ delta,  // delta (tok, D_INNER)
                            const float* __restrict__ xdbl,   // (tok, 96): [dt_r | B | C]
                            const float* __restrict__ A_log,  // (D_INNER, D_STATE)
                            const float* __restrict__ Dvec,   // (D_INNER)
                            float* __restrict__ y)            // (tok, D_INNER)
{
    const int tid = blockIdx.x * blockDim.x + threadIdx.x;
    const int n   = tid & 15;
    const int ch  = tid >> 4;
    if (ch >= BATCH * D_INNER) return;
    const int d = ch % D_INNER;
    const int b = ch / D_INNER;

    const float A  = -__expf(A_log[d * D_STATE + n]);
    const float Dd = Dvec[d];

    float state = 0.0f;
    for (int t = 0; t < SEQLEN; ++t) {
        const size_t tok = (size_t)b * SEQLEN + t;
        const float dt = delta[tok * D_INNER + d];
        const float u  = xc[tok * D_INNER + d];
        const float Bv = xdbl[tok * XPROJ_N + DT_RANK + n];
        const float Cv = xdbl[tok * XPROJ_N + DT_RANK + D_STATE + n];

        state = state * __expf(dt * A) + (dt * u) * Bv;

        float part = state * Cv;
        part += __shfl_xor(part, 8, 16);
        part += __shfl_xor(part, 4, 16);
        part += __shfl_xor(part, 2, 16);
        part += __shfl_xor(part, 1, 16);
        if (n == 0)
            y[tok * D_INNER + d] = part + Dd * u;
    }
}

// -----------------------------------------------------------------------------
// Gating: yb = y * silu(z) as bf16 (input to out_proj GEMM)
// -----------------------------------------------------------------------------
__global__ void gate_kernel(const float* __restrict__ xz,
                            const float* __restrict__ y,
                            __bf16* __restrict__ y_bf)
{
    const int idx = blockIdx.x * blockDim.x + threadIdx.x;
    if (idx >= NTOK * D_INNER) return;
    const int d   = idx % D_INNER;
    const int tok = idx / D_INNER;
    const float z = xz[(size_t)tok * (2 * D_INNER) + D_INNER + d];
    y_bf[(size_t)tok * D_INNER + d] = (__bf16)(y[(size_t)tok * D_INNER + d] * silu_f(z));
}

// -----------------------------------------------------------------------------
extern "C" void kernel_launch(void* const* d_in, const int* in_sizes, int n_in,
                              void* d_out, int out_size, void* d_ws, size_t ws_size,
                              hipStream_t stream)
{
    const float* hidden    = (const float*)d_in[0]; // (NTOK, D_MODEL)
    const float* in_proj_w = (const float*)d_in[1]; // (2*D_INNER, D_MODEL)
    const float* conv_w    = (const float*)d_in[2];
    const float* conv_b    = (const float*)d_in[3];
    const float* x_proj_w  = (const float*)d_in[4]; // (96, D_INNER)
    const float* dt_proj_w = (const float*)d_in[5]; // (D_INNER, DT_RANK)
    const float* dt_proj_b = (const float*)d_in[6];
    const float* A_log     = (const float*)d_in[7];
    const float* Dvec      = (const float*)d_in[8];
    const float* out_w     = (const float*)d_in[9]; // (D_MODEL, D_INNER)
    float* out = (float*)d_out;                     // (NTOK, D_MODEL)

    // ---- workspace layout ----
    char* ws = (char*)d_ws;
    float* xz    = (float*)ws;  ws += sizeof(float) * (size_t)NTOK * 2 * D_INNER;
    float* xc    = (float*)ws;  ws += sizeof(float) * (size_t)NTOK * D_INNER;
    float* xdbl  = (float*)ws;  ws += sizeof(float) * (size_t)NTOK * XPROJ_N;
    float* delta = (float*)ws;  ws += sizeof(float) * (size_t)NTOK * D_INNER;
    float* y     = (float*)ws;  ws += sizeof(float) * (size_t)NTOK * D_INNER;
    __bf16* hidden_bf = (__bf16*)ws; ws += sizeof(__bf16) * (size_t)NTOK * D_MODEL;
    __bf16* in_w_bf   = (__bf16*)ws; ws += sizeof(__bf16) * (size_t)2 * D_INNER * D_MODEL;
    __bf16* xp_w_bf   = (__bf16*)ws; ws += sizeof(__bf16) * (size_t)XPROJ_N * D_INNER;
    __bf16* dt_w_bf   = (__bf16*)ws; ws += sizeof(__bf16) * (size_t)D_INNER * DT_RANK;
    __bf16* out_w_bf  = (__bf16*)ws; ws += sizeof(__bf16) * (size_t)D_MODEL * D_INNER;
    __bf16* xc_bf     = (__bf16*)ws; ws += sizeof(__bf16) * (size_t)NTOK * D_INNER;
    __bf16* xdbl_bf   = (__bf16*)ws; ws += sizeof(__bf16) * (size_t)NTOK * XPROJ_N;
    __bf16* y_bf      = (__bf16*)ws;

    const dim3 blk128(128, 1, 1);
    const dim3 blk256(256, 1, 1);
    auto cvtgrid = [](size_t n) { return dim3((unsigned)((n / 4 + 255) / 256), 1, 1); };

    // ---- 0) one-time bf16 conversions (input + weights) ----
    cvt_f32_bf16<<<cvtgrid((size_t)NTOK * D_MODEL), blk256, 0, stream>>>(hidden, hidden_bf, NTOK * D_MODEL);
    cvt_f32_bf16<<<cvtgrid((size_t)2 * D_INNER * D_MODEL), blk256, 0, stream>>>(in_proj_w, in_w_bf, 2 * D_INNER * D_MODEL);
    cvt_f32_bf16<<<cvtgrid((size_t)XPROJ_N * D_INNER), blk256, 0, stream>>>(x_proj_w, xp_w_bf, XPROJ_N * D_INNER);
    cvt_f32_bf16<<<cvtgrid((size_t)D_INNER * DT_RANK), blk256, 0, stream>>>(dt_proj_w, dt_w_bf, D_INNER * DT_RANK);
    cvt_f32_bf16<<<cvtgrid((size_t)D_MODEL * D_INNER), blk256, 0, stream>>>(out_w, out_w_bf, D_MODEL * D_INNER);

    // ---- 1) in_proj: xz = hidden @ in_proj_w^T  (M=4096, N=4096, K=1024)
    gemm_bf16_wmma<4, 0><<<dim3((2 * D_INNER) / 64, NTOK / 64), blk128, 0, stream>>>(
        hidden_bf, in_w_bf, nullptr, xz, nullptr,
        D_MODEL, D_MODEL, D_MODEL, 2 * D_INNER, 0);

    // ---- 2) conv + SiLU -> xc (f32 + bf16)
    conv_silu_kernel<<<(NTOK * D_INNER + 255) / 256, blk256, 0, stream>>>(
        xz, conv_w, conv_b, xc, xc_bf);

    // ---- 3) x_proj: xdbl = xc @ x_proj_w^T  (N=96 -> 64 + 32 remainder)
    gemm_bf16_wmma<4, 0><<<dim3(1, NTOK / 64), blk128, 0, stream>>>(
        xc_bf, xp_w_bf, nullptr, xdbl, xdbl_bf,
        D_INNER, D_INNER, D_INNER, XPROJ_N, 0);
    gemm_bf16_wmma<2, 0><<<dim3(1, NTOK / 64), blk128, 0, stream>>>(
        xc_bf, xp_w_bf, nullptr, xdbl, xdbl_bf,
        D_INNER, D_INNER, D_INNER, XPROJ_N, 64);

    // ---- 4) dt_proj + bias + softplus -> delta  (K=64 from xdbl cols 0..63)
    gemm_bf16_wmma<4, 1><<<dim3(D_INNER / 64, NTOK / 64), blk128, 0, stream>>>(
        xdbl_bf, dt_w_bf, dt_proj_b, delta, nullptr,
        DT_RANK, XPROJ_N, DT_RANK, D_INNER, 0);

    // ---- 5) selective scan -> y
    scan_kernel<<<(BATCH * D_INNER * D_STATE) / 256, blk256, 0, stream>>>(
        xc, delta, xdbl, A_log, Dvec, y);

    // ---- 6) gating -> y_bf
    gate_kernel<<<(NTOK * D_INNER + 255) / 256, blk256, 0, stream>>>(xz, y, y_bf);

    // ---- 7) out_proj: out = y_gated @ out_proj_w^T
    gemm_bf16_wmma<4, 0><<<dim3(D_MODEL / 64, NTOK / 64), blk128, 0, stream>>>(
        y_bf, out_w_bf, nullptr, out, nullptr,
        D_INNER, D_INNER, D_INNER, D_MODEL, 0);
}
